// ProductOfGaussians_63986422776343
// MI455X (gfx1250) — matrix operand order repaired and checked by reading, not compiled, not run on hardware
//
#include <hip/hip_runtime.h>
#include <math.h>

#define T_STEPS 8192
#define DZ 32

typedef __attribute__((ext_vector_type(16))) __bf16 v16bf;
typedef __attribute__((ext_vector_type(8)))  float  v8f;

// ---------------------------------------------------------------------------
// fp32 -> bf16 (round-to-nearest-even) elementwise convert
// ---------------------------------------------------------------------------
__global__ void f32_to_bf16_kernel(const float* __restrict__ in,
                                   unsigned short* __restrict__ out, int n) {
  int i = blockIdx.x * blockDim.x + threadIdx.x;
  if (i < n) {
    unsigned int u = __float_as_uint(in[i]);
    u += 0x7FFFu + ((u >> 16) & 1u);
    out[i] = (unsigned short)(u >> 16);
  }
}

// ---------------------------------------------------------------------------
// CDNA5 async global->LDS copy of one 64-byte row (4 x B128).
// IOFFSET is added to BOTH the LDS address and the global address (ISA 15.18.3),
// so a single LDS-offset VGPR + immediate offsets covers the row.
// Tracked by ASYNCcnt.
// ---------------------------------------------------------------------------
__device__ __forceinline__ void async_row64(unsigned ldsOff,
                                            const unsigned short* gptr) {
  unsigned long long ga = (unsigned long long)(size_t)gptr;
  asm volatile(
      "global_load_async_to_lds_b128 %0, %1, off offset:0\n\t"
      "global_load_async_to_lds_b128 %0, %1, off offset:16\n\t"
      "global_load_async_to_lds_b128 %0, %1, off offset:32\n\t"
      "global_load_async_to_lds_b128 %0, %1, off offset:48"
      :: "v"(ldsOff), "v"(ga) : "memory");
}

__device__ __forceinline__ void wait_async0() {
  asm volatile("s_wait_asynccnt 0" ::: "memory");
}

// ---------------------------------------------------------------------------
// WMMA GEMM: out[M,N] = act(A[M,K](bf16) @ W[N,K](bf16)^T + bias[N])
// M fixed at 8192. BM=BN=128, BK=32; 128 threads = 4 waves (2x2),
// each wave computes a 64x64 tile via 4x4 v_wmma_f32_16x16x32_bf16
// (16 WMMA : 16 ds_load_b128 per K-step). LDS tiles are double-buffered and
// filled with global_load_async_to_lds_b128; one barrier per K-step.
// ---------------------------------------------------------------------------
union FragBF16 { uint4 u[2]; v16bf v; };

__launch_bounds__(128)
__global__ void wmma_gemm_kernel(const unsigned short* __restrict__ Abf,
                                 const unsigned short* __restrict__ Wbf,
                                 const float* __restrict__ bias,
                                 float* __restrict__ outF,
                                 unsigned short* __restrict__ outB,
                                 int N, int K, int relu, int store_bf16) {
  __shared__ __align__(16) unsigned short Ash[2][128][40]; // 80B row stride
  __shared__ __align__(16) unsigned short Bsh[2][128][40];

  const int tid = threadIdx.x;
  const int nb = (N + 127) >> 7;
  const int blockN = blockIdx.x % nb;
  const int blockM = blockIdx.x / nb;
  const int wid = tid >> 5, lane = tid & 31;
  const int waveM = wid >> 1, waveN = wid & 1;
  const int halfId = lane >> 4, ln = lane & 15;
  const int row = tid; // 0..127: one staged row per thread

  const unsigned short* Arow = Abf + (size_t)(blockM * 128 + row) * K;
  const int gn = blockN * 128 + row;
  const bool bvalid = (gn < N);
  const unsigned short* Brow = Wbf + (size_t)(bvalid ? gn : 0) * K;

  unsigned aLds[2], bLds[2];
  aLds[0] = (unsigned)(size_t)&Ash[0][row][0];
  aLds[1] = (unsigned)(size_t)&Ash[1][row][0];
  bLds[0] = (unsigned)(size_t)&Bsh[0][row][0];
  bLds[1] = (unsigned)(size_t)&Bsh[1][row][0];

  // Zero B rows that have no valid weight row (stay zero: never async-written).
  if (!bvalid) {
    uint4 z = make_uint4(0u, 0u, 0u, 0u);
#pragma unroll
    for (int i = 0; i < 4; i++) {
      *reinterpret_cast<uint4*>(&Bsh[0][row][i * 8]) = z;
      *reinterpret_cast<uint4*>(&Bsh[1][row][i * 8]) = z;
    }
  }

  // Prologue: async-stage tile 0 into buffer 0.
  async_row64(aLds[0], Arow);
  if (bvalid) async_row64(bLds[0], Brow);
  wait_async0();
  __syncthreads();

  v8f acc[4][4] = {};
  const int S = K >> 5;
  int cur = 0;

  for (int s = 0; s < S; ++s) {
    // Issue async copies for the next K-tile into the other buffer; they
    // overlap with the 16 WMMAs below.
    if (s + 1 < S) {
      int kt = (s + 1) << 5;
      async_row64(aLds[cur ^ 1], Arow + kt);
      if (bvalid) async_row64(bLds[cur ^ 1], Brow + kt);
    }

    // Preload all four B fragments, then stream A fragments.
    FragBF16 b[4];
#pragma unroll
    for (int tn = 0; tn < 4; tn++) {
      int c = waveN * 64 + tn * 16 + ln;
      // B layout: half 0 -> K 0..7 & 16..23; half 1 -> K 8..15 & 24..31 of col c
      b[tn].u[0] = *reinterpret_cast<const uint4*>(&Bsh[cur][c][8 * halfId]);
      b[tn].u[1] = *reinterpret_cast<const uint4*>(&Bsh[cur][c][8 * halfId + 16]);
    }
#pragma unroll
    for (int tm = 0; tm < 4; tm++) {
      int r = waveM * 64 + tm * 16 + ln;
      FragBF16 a;
      // 16-bit A layout: half 0 -> K 0..7 & 16..23; half 1 -> K 8..15 & 24..31
      a.u[0] = *reinterpret_cast<const uint4*>(&Ash[cur][r][8 * halfId]);
      a.u[1] = *reinterpret_cast<const uint4*>(&Ash[cur][r][8 * halfId + 16]);
#pragma unroll
      for (int tn = 0; tn < 4; tn++)
        acc[tm][tn] = __builtin_amdgcn_wmma_f32_16x16x32_bf16(
            false, a.v, false, b[tn].v, (short)0, acc[tm][tn], false, false);
    }

    wait_async0();     // next tile fully resident in LDS (this wave's part)
    __syncthreads();   // all waves done computing + staging
    cur ^= 1;
  }

  // Epilogue: C layout — VGPR v holds M = v + 8*halfId, lane%16 gives N.
#pragma unroll
  for (int tm = 0; tm < 4; tm++)
#pragma unroll
    for (int tn = 0; tn < 4; tn++)
#pragma unroll
      for (int v = 0; v < 8; v++) {
        int m = blockM * 128 + waveM * 64 + tm * 16 + halfId * 8 + v;
        int n = blockN * 128 + waveN * 64 + tn * 16 + ln;
        if (n < N) {
          float val = acc[tm][tn][v] + bias[n];
          if (relu) val = fmaxf(val, 0.0f);
          if (store_bf16) {
            unsigned int u = __float_as_uint(val);
            u += 0x7FFFu + ((u >> 16) & 1u);
            outB[(size_t)m * N + n] = (unsigned short)(u >> 16);
          } else {
            outF[(size_t)m * N + n] = val;
          }
        }
      }
}

// ---------------------------------------------------------------------------
// Prior matrices: Qinv = Qc Qc^T, Q0inv = Q0c Q0c^T, AQA = A^T Qinv A,
// P0 = Q0inv+AQA, Pmid = Qinv+AQA, Pend = Qinv, Boff = -(A^T Qinv).
// ---------------------------------------------------------------------------
__launch_bounds__(1024)
__global__ void priors_kernel(const float* __restrict__ A,
                              const float* __restrict__ Qc,
                              const float* __restrict__ Q0c,
                              float* __restrict__ P0, float* __restrict__ Pmid,
                              float* __restrict__ Pend, float* __restrict__ Boff) {
  __shared__ float As[1024], Qs[1024], Q0s[1024], Qinv[1024], Tmp[1024];
  int t = threadIdx.x;
  As[t] = A[t]; Qs[t] = Qc[t]; Q0s[t] = Q0c[t];
  __syncthreads();
  int i = t >> 5, j = t & 31;
  float s = 0.f, s0 = 0.f;
  for (int k = 0; k < 32; k++) {
    s  += Qs[i * 32 + k]  * Qs[j * 32 + k];
    s0 += Q0s[i * 32 + k] * Q0s[j * 32 + k];
  }
  Qinv[t] = s;
  float q0inv = s0;
  __syncthreads();
  float tv = 0.f;
  for (int k = 0; k < 32; k++) tv += Qinv[i * 32 + k] * As[k * 32 + j];
  Tmp[t] = tv;
  __syncthreads();
  float aqa = 0.f, bo = 0.f;
  for (int k = 0; k < 32; k++) {
    aqa += As[k * 32 + i] * Tmp[k * 32 + j];
    bo  += As[k * 32 + i] * Qinv[k * 32 + j];
  }
  P0[t]   = q0inv + aqa;
  Pmid[t] = Qinv[t] + aqa;
  Pend[t] = Qinv[t];
  Boff[t] = -bo;
}

// ---------------------------------------------------------------------------
// Per-t: Lam = R R^T (R = cov row reshaped 32x32), AA = Lam + prior (in place
// over the cov buffer), lamMu = Lam @ mean. One block per t, 256 threads.
// ---------------------------------------------------------------------------
__launch_bounds__(256)
__global__ void build_AA_kernel(float* __restrict__ covAA,
                                const float* __restrict__ mean,
                                const float* __restrict__ P0,
                                const float* __restrict__ Pmid,
                                const float* __restrict__ Pend,
                                float* __restrict__ lamMu) {
  __shared__ float covS[1024];
  __shared__ float LamS[1024];
  __shared__ float meanS[32];
  int t = blockIdx.x, tid = threadIdx.x;
  const float* cv = covAA + (size_t)t * 1024;
  for (int e = tid; e < 1024; e += 256) covS[e] = cv[e];
  if (tid < 32) meanS[tid] = mean[(size_t)t * 32 + tid];
  __syncthreads();
  const float* Pr = (t == 0) ? P0 : ((t == T_STEPS - 1) ? Pend : Pmid);
  float outv[4];
#pragma unroll
  for (int q = 0; q < 4; q++) {
    int e = tid + q * 256;
    int i = e >> 5, j = e & 31;
    float s = 0.f;
    for (int k = 0; k < 32; k++) s += covS[i * 32 + k] * covS[j * 32 + k];
    LamS[e] = s;
    outv[q] = s + Pr[e];
  }
  __syncthreads();
#pragma unroll
  for (int q = 0; q < 4; q++) covAA[(size_t)t * 1024 + tid + q * 256] = outv[q];
  if (tid < 32) {
    float s = 0.f;
    for (int j = 0; j < 32; j++) s += LamS[tid * 32 + j] * meanS[j];
    lamMu[(size_t)t * 32 + tid] = s;
  }
}

// ---------------------------------------------------------------------------
// Sequential block-tridiagonal Cholesky scan. Single wave32 workgroup:
// barrier ops are hardware NOPs, LDS is in-order within the wave.
// ---------------------------------------------------------------------------
__launch_bounds__(32)
__global__ void scan_chol_kernel(float* __restrict__ AAR,
                                 float* __restrict__ Xg,
                                 const float* __restrict__ Boff,
                                 float* __restrict__ entropy_out) {
  __shared__ float Lsh[32][33];
  __shared__ float Xsh[32][33];
  __shared__ float Bsh[32][33];
  __shared__ float Ash[32][33];
  const int lane = threadIdx.x;
  for (int k = 0; k < 32; k++) Bsh[k][lane] = Boff[k * 32 + lane];
  __syncthreads();
  float logacc = 0.f;

  for (int t = 0; t < T_STEPS; t++) {
    const float* AAt = AAR + (size_t)t * 1024;
    for (int k = 0; k < 32; k++) Ash[k][lane] = AAt[k * 32 + lane];
    __syncthreads();

    if (t > 0) {
      // Forward substitution: column `lane` of X = Lprev^{-1} B.
      float xc[32];
#pragma unroll
      for (int i = 0; i < 32; i++) {
        float s = Bsh[i][lane];
#pragma unroll
        for (int k2 = 0; k2 < i; k2++) s -= Lsh[i][k2] * xc[k2];
        xc[i] = s / Lsh[i][i];
      }
#pragma unroll
      for (int i = 0; i < 32; i++) Xsh[i][lane] = xc[i];
      __syncthreads();
      float* Xt = Xg + (size_t)(t - 1) * 1024;
      for (int i = 0; i < 32; i++) Xt[i * 32 + lane] = xc[i];
      // M = AA - X^T X  (column `lane`)
      for (int i = 0; i < 32; i++) {
        float s = Ash[i][lane];
#pragma unroll
        for (int k2 = 0; k2 < 32; k2++) s -= Xsh[k2][i] * xc[k2];
        Ash[i][lane] = s;
      }
      __syncthreads();
    }

    // Cholesky of Ash (lower) -> Lsh; lane = row.
    for (int j = 0; j < 32; j++) {
      float d = sqrtf(Ash[j][j]);
      float inv = 1.0f / d;
      float lij = (lane == j) ? d : ((lane > j) ? Ash[lane][j] * inv : 0.0f);
      Lsh[lane][j] = lij;
      if (lane == j) logacc += logf(d);
      __syncthreads();
      for (int k2 = j + 1; k2 <= lane; k2++) Ash[lane][k2] -= lij * Lsh[k2][j];
      __syncthreads();
    }
    float* Rt = AAR + (size_t)t * 1024;
    for (int k = 0; k < 32; k++) Rt[k * 32 + lane] = Lsh[k][lane];
    __syncthreads();
  }

  for (int offi = 16; offi > 0; offi >>= 1) logacc += __shfl_down(logacc, offi, 32);
  if (lane == 0) {
    float ln_det = -2.0f * logacc;
    float two_pi = 6.2831853071795864769f;
    float ent_inner = 0.5f * ln_det + (32.0f * 50.0f * 0.5f) * (1.0f + logf(two_pi));
    entropy_out[0] = (-ent_inner / 50.0f) / (float)T_STEPS;
  }
}

// ---------------------------------------------------------------------------
// Bidiagonal block solves (single wave32). Forward: ib. Backward: mu_z and
// noise simultaneously; sample = mu_z + noise -> d_out.
// ---------------------------------------------------------------------------
__launch_bounds__(32)
__global__ void solve_kernel(const float* __restrict__ Rg,
                             const float* __restrict__ Xg,
                             const float* __restrict__ lamMu,
                             const float* __restrict__ norm,
                             float* __restrict__ ib,
                             float* __restrict__ sample) {
  __shared__ float Lsh[32][33];
  __shared__ float Csh[32][33];
  __shared__ float xp[32], x1s[32], x2s[32];
  const int lane = threadIdx.x;

  for (int t = 0; t < T_STEPS; t++) {
    const float* Rt = Rg + (size_t)t * 1024;
    for (int k = 0; k < 32; k++) Lsh[k][lane] = Rt[k * 32 + lane];
    float b = lamMu[(size_t)t * 32 + lane];
    if (t > 0) {
      const float* Xt = Xg + (size_t)(t - 1) * 1024;
      for (int k = 0; k < 32; k++) Csh[k][lane] = Xt[k * 32 + lane];
      __syncthreads();
      for (int k = 0; k < 32; k++) b -= Csh[k][lane] * xp[k];
    }
    __syncthreads();
    float acc = b, x = 0.f;
    for (int j = 0; j < 32; j++) {
      float tj = __shfl(acc, j, 32);
      float xj = tj / Lsh[j][j];
      if (lane == j) x = xj;
      if (lane > j) acc -= Lsh[lane][j] * xj;
    }
    ib[(size_t)t * 32 + lane] = x;
    __syncthreads();
    xp[lane] = x;
    __syncthreads();
  }

  for (int t = T_STEPS - 1; t >= 0; t--) {
    const float* Rt = Rg + (size_t)t * 1024;
    for (int k = 0; k < 32; k++) Lsh[k][lane] = Rt[k * 32 + lane];
    float b1 = ib[(size_t)t * 32 + lane];
    float b2 = norm[(size_t)t * 32 + lane];
    if (t < T_STEPS - 1) {
      const float* Xt = Xg + (size_t)t * 1024;
      for (int k = 0; k < 32; k++) Csh[k][lane] = Xt[k * 32 + lane];
      __syncthreads();
      for (int k = 0; k < 32; k++) {
        float c = Csh[lane][k];
        b1 -= c * x1s[k];
        b2 -= c * x2s[k];
      }
    }
    __syncthreads();
    float a1 = b1, a2 = b2, x1 = 0.f, x2 = 0.f;
    for (int j = 31; j >= 0; j--) {
      float t1 = __shfl(a1, j, 32);
      float t2 = __shfl(a2, j, 32);
      float d = Lsh[j][j];
      float xj1 = t1 / d, xj2 = t2 / d;
      if (lane == j) { x1 = xj1; x2 = xj2; }
      if (lane < j) {
        float lj = Lsh[j][lane];   // U[lane][j] = L[j][lane]
        a1 -= lj * xj1;
        a2 -= lj * xj2;
      }
    }
    sample[(size_t)t * 32 + lane] = x1 + x2;
    __syncthreads();
    x1s[lane] = x1;
    x2s[lane] = x2;
    __syncthreads();
  }
}

// ---------------------------------------------------------------------------
// Host launch
// ---------------------------------------------------------------------------
extern "C" void kernel_launch(void* const* d_in, const int* in_sizes, int n_in,
                              void* d_out, int out_size, void* d_ws, size_t ws_size,
                              hipStream_t stream) {
  const float* x     = (const float*)d_in[0];
  const float* nrm   = (const float*)d_in[1];
  const float* A     = (const float*)d_in[2];
  const float* Qc    = (const float*)d_in[3];
  const float* Q0c   = (const float*)d_in[4];
  const float* Wm_in = (const float*)d_in[5];  const float* bm_in = (const float*)d_in[6];
  const float* Wm_h1 = (const float*)d_in[7];  const float* bm_h1 = (const float*)d_in[8];
  const float* Wm_h3 = (const float*)d_in[9];  const float* bm_h3 = (const float*)d_in[10];
  const float* Wm_out= (const float*)d_in[11]; const float* bm_out= (const float*)d_in[12];
  const float* Wc_in = (const float*)d_in[13]; const float* bc_in = (const float*)d_in[14];
  const float* Wc_h1 = (const float*)d_in[15]; const float* bc_h1 = (const float*)d_in[16];
  const float* Wc_h3 = (const float*)d_in[17]; const float* bc_h3 = (const float*)d_in[18];
  const float* Wc_out= (const float*)d_in[19]; const float* bc_out= (const float*)d_in[20];

  char* ws = (char*)d_ws;
  size_t off = 0;
  auto carve = [&](size_t bytes) -> char* {
    char* p = ws + off;
    off += (bytes + 255) & ~(size_t)255;
    return p;
  };

  unsigned short* xb    = (unsigned short*)carve((size_t)8192 * 256 * 2);
  unsigned short* wmin  = (unsigned short*)carve((size_t)1024 * 256 * 2);
  unsigned short* wmh1  = (unsigned short*)carve((size_t)1024 * 1024 * 2);
  unsigned short* wmh3  = (unsigned short*)carve((size_t)1024 * 1024 * 2);
  unsigned short* wmout = (unsigned short*)carve((size_t)32 * 1024 * 2);
  unsigned short* wcin  = (unsigned short*)carve((size_t)1024 * 256 * 2);
  unsigned short* wch1  = (unsigned short*)carve((size_t)1024 * 1024 * 2);
  unsigned short* wch3  = (unsigned short*)carve((size_t)1024 * 1024 * 2);
  unsigned short* wcout = (unsigned short*)carve((size_t)1024 * 1024 * 2);
  unsigned short* hA    = (unsigned short*)carve((size_t)8192 * 1024 * 2);
  unsigned short* hB    = (unsigned short*)carve((size_t)8192 * 1024 * 2);
  float* meanF  = (float*)carve((size_t)8192 * 32 * 4);
  float* covAA  = (float*)carve((size_t)8192 * 1024 * 4);  // cov -> AA -> R (in place)
  float* Xg     = (float*)carve((size_t)8192 * 1024 * 4);
  float* lamMu  = (float*)carve((size_t)8192 * 32 * 4);
  float* ibuf   = (float*)carve((size_t)8192 * 32 * 4);
  float* P0     = (float*)carve(1024 * 4);
  float* Pmid   = (float*)carve(1024 * 4);
  float* Pend   = (float*)carve(1024 * 4);
  float* Boff   = (float*)carve(1024 * 4);

  auto conv = [&](const float* src, unsigned short* dst, int n) {
    f32_to_bf16_kernel<<<dim3((n + 255) / 256), dim3(256), 0, stream>>>(src, dst, n);
  };
  conv(x, xb, 8192 * 256);
  conv(Wm_in, wmin, 1024 * 256);
  conv(Wm_h1, wmh1, 1024 * 1024);
  conv(Wm_h3, wmh3, 1024 * 1024);
  conv(Wm_out, wmout, 32 * 1024);
  conv(Wc_in, wcin, 1024 * 256);
  conv(Wc_h1, wch1, 1024 * 1024);
  conv(Wc_h3, wch3, 1024 * 1024);
  conv(Wc_out, wcout, 1024 * 1024);

  auto gemm = [&](const unsigned short* Ain, const unsigned short* W, const float* bias,
                  float* oF, unsigned short* oB, int N, int K, int relu, int sbf) {
    int nb = (N + 127) / 128;
    wmma_gemm_kernel<<<dim3(nb * (8192 / 128)), dim3(128), 0, stream>>>(
        Ain, W, bias, oF, oB, N, K, relu, sbf);
  };

  // mean MLP
  gemm(xb, wmin, bm_in, nullptr, hA, 1024, 256, 1, 1);
  gemm(hA, wmh1, bm_h1, nullptr, hB, 1024, 1024, 1, 1);
  gemm(hB, wmh3, bm_h3, nullptr, hA, 1024, 1024, 1, 1);
  gemm(hA, wmout, bm_out, meanF, nullptr, 32, 1024, 0, 0);
  // cov MLP
  gemm(xb, wcin, bc_in, nullptr, hA, 1024, 256, 1, 1);
  gemm(hA, wch1, bc_h1, nullptr, hB, 1024, 1024, 1, 1);
  gemm(hB, wch3, bc_h3, nullptr, hA, 1024, 1024, 1, 1);
  gemm(hA, wcout, bc_out, covAA, nullptr, 1024, 1024, 0, 0);

  priors_kernel<<<dim3(1), dim3(1024), 0, stream>>>(A, Qc, Q0c, P0, Pmid, Pend, Boff);
  build_AA_kernel<<<dim3(T_STEPS), dim3(256), 0, stream>>>(covAA, meanF, P0, Pmid, Pend, lamMu);

  float* out_sample  = (float*)d_out;
  float* out_entropy = out_sample + (size_t)T_STEPS * DZ;
  scan_chol_kernel<<<dim3(1), dim3(32), 0, stream>>>(covAA, Xg, Boff, out_entropy);
  solve_kernel<<<dim3(1), dim3(32), 0, stream>>>(covAA, Xg, lamMu, nrm, ibuf, out_sample);
}
